// factor_updating_structure_7610682049159
// MI455X (gfx1250) — compile-verified, matrix operand order checked by reading, not run on hardware
//
#include <hip/hip_runtime.h>

typedef __attribute__((ext_vector_type(16))) __bf16        v16bf;
typedef __attribute__((ext_vector_type(8)))  float         v8f;
typedef __attribute__((ext_vector_type(8)))  unsigned int  v8u;

#define N_OBJ_C 4096
#define N_REG_C 8192
#define D_C     512
#define GW_C    128
#define E_C     131072

__device__ __forceinline__ unsigned short f2bf(float x) {
    unsigned int u = __builtin_bit_cast(unsigned int, x);
    unsigned int r = u + 0x7FFFu + ((u >> 16) & 1u);   // round-to-nearest-even
    return (unsigned short)(r >> 16);
}
__device__ __forceinline__ unsigned int pack2bf(float lo, float hi) {
    return (unsigned int)f2bf(lo) | ((unsigned int)f2bf(hi) << 16);
}

// ---------------------------------------------------------------------------
// P1: relu + fp32->bf16 convert (features used as gate-GEMM A operands)
// ---------------------------------------------------------------------------
__global__ __launch_bounds__(256) void relu_bf16_kernel(
    const float* __restrict__ in, unsigned short* __restrict__ out, int n) {
    int i = blockIdx.x * 256 + threadIdx.x;
    if (i < n) out[i] = f2bf(fmaxf(in[i], 0.0f));
}

// ---------------------------------------------------------------------------
// P2: pack weight W[N,Ktot] (row-major) into WMMA bf16 B-fragment layout:
//     pack[((kk*Ntiles + t)*32 + lane)*8 + p] = bf16pair( W[n, k], W[n, k+1] )
//     n = t*16 + (lane&15);  k = kk*32 + (p<4 ? hi*8+2p : 16+hi*8+2(p-4))
// ---------------------------------------------------------------------------
__global__ __launch_bounds__(256) void pack_w_kernel(
    const float* __restrict__ w, unsigned int* __restrict__ pack,
    int Ktot, int Ntiles, int total) {
    int id = blockIdx.x * 256 + threadIdx.x;
    if (id >= total) return;
    int p    = id & 7;
    int lane = (id >> 3) & 31;
    int rest = id >> 8;
    int t    = rest % Ntiles;
    int kk   = rest / Ntiles;
    int hi   = lane >> 4;
    int n    = t * 16 + (lane & 15);
    int k    = kk * 32 + ((p < 4) ? (hi * 8 + 2 * p) : (16 + hi * 8 + 2 * (p - 4)));
    pack[id] = pack2bf(w[(long)n * Ktot + k], w[(long)n * Ktot + k + 1]);
}

// ---------------------------------------------------------------------------
// K1: gate kernel. One wave per 16-edge tile.
//     A = relu(cat(src,tgt)) gathered bf16 rows, B = packed gate weights.
//     gate[e] = mean_n sigmoid( (A x B)[e,n] + gb[n] )
// ---------------------------------------------------------------------------
__global__ __launch_bounds__(256) void gate_kernel(
    const unsigned short* __restrict__ reluSrc,   // [nSrc, 512] bf16 (relu'd)
    const unsigned short* __restrict__ reluTgt,   // [nTgt, 512] bf16 (relu'd)
    const int* __restrict__ srcIdx,
    const int* __restrict__ tgtIdx,
    const unsigned int* __restrict__ bpack,       // [32][8][32][8] dwords
    const float* __restrict__ gb,                 // [128]
    float* __restrict__ gateOut,                  // [E]
    int nEdges) {
    int lane = threadIdx.x & 31;
    int wave = threadIdx.x >> 5;
    int tile = blockIdx.x * 8 + wave;
    int e0   = tile * 16;
    if (e0 >= nEdges) return;                     // wave-uniform guard

    int r  = lane & 15;
    int hi = lane >> 4;
    int e  = e0 + r;
    const unsigned short* srow = reluSrc + (long)srcIdx[e] * D_C;
    const unsigned short* trow = reluTgt + (long)tgtIdx[e] * D_C;

    v8f zero = {0.f, 0.f, 0.f, 0.f, 0.f, 0.f, 0.f, 0.f};
    v8f c[8];
#pragma unroll
    for (int t = 0; t < 8; ++t) c[t] = zero;

    for (int kk = 0; kk < 32; ++kk) {
        int K0 = kk * 32;
        const unsigned short* rowb = (K0 < D_C) ? srow : trow;
        int kb = K0 & (D_C - 1);
        v8u au;
#pragma unroll
        for (int p = 0; p < 4; ++p)
            au[p] = *(const unsigned int*)(rowb + kb + hi * 8 + 2 * p);
#pragma unroll
        for (int p = 0; p < 4; ++p)
            au[4 + p] = *(const unsigned int*)(rowb + kb + 16 + hi * 8 + 2 * p);
        v16bf a = __builtin_bit_cast(v16bf, au);

        const v8u* bkk = (const v8u*)bpack + (long)(kk * 8) * 32 + lane;
#pragma unroll
        for (int t = 0; t < 8; ++t) {
            v16bf b = __builtin_bit_cast(v16bf, bkk[t * 32]);
            c[t] = __builtin_amdgcn_wmma_f32_16x16x32_bf16(
                false, a, false, b, (short)0, c[t], false, false);
        }
    }

    // sigmoid + mean over 128 gate columns
    float rowsum[8];
#pragma unroll
    for (int q = 0; q < 8; ++q) rowsum[q] = 0.f;
#pragma unroll
    for (int t = 0; t < 8; ++t) {
        float bias = gb[t * 16 + r];
#pragma unroll
        for (int q = 0; q < 8; ++q) {
            float x = c[t][q] + bias;
            rowsum[q] += 1.0f / (1.0f + __expf(-x));
        }
    }
    // reduce across the 16 column lanes of each half-wave
#pragma unroll
    for (int m = 1; m < 16; m <<= 1) {
#pragma unroll
        for (int q = 0; q < 8; ++q) rowsum[q] += __shfl_xor(rowsum[q], m, 32);
    }
    if (r == 0) {
        int base = e0 + hi * 8;   // lane0: rows 0..7, lane16: rows 8..15
#pragma unroll
        for (int q = 0; q < 8; ++q)
            gateOut[base + q] = rowsum[q] * (1.0f / 128.0f);
    }
}

// ---------------------------------------------------------------------------
// K2: segmented mean (tgt sorted) + relu + bf16. One block per target.
// ---------------------------------------------------------------------------
__device__ __forceinline__ int lower_bound_dev(const int* __restrict__ a, int n, int key) {
    int lo = 0, hi = n;
    while (lo < hi) {
        int mid = (lo + hi) >> 1;
        if (a[mid] < key) lo = mid + 1; else hi = mid;
    }
    return lo;
}

__global__ __launch_bounds__(128) void agg_kernel(
    const float* __restrict__ srcFeat,            // [nSrc, 512] fp32 (raw)
    const int* __restrict__ srcIdx,
    const int* __restrict__ tgtSorted,
    const float* __restrict__ gate,
    unsigned short* __restrict__ mreluOut,        // [nTgt, 512] bf16
    int nEdges) {
    int t  = blockIdx.x;
    int lo = lower_bound_dev(tgtSorted, nEdges, t);
    int hi = lower_bound_dev(tgtSorted, nEdges, t + 1);
    int d  = threadIdx.x * 4;

    float4 acc = {0.f, 0.f, 0.f, 0.f};
    for (int e = lo; e < hi; ++e) {
        float g = gate[e];
        const float* srow = srcFeat + (long)srcIdx[e] * D_C;
        float4 v = *(const float4*)(srow + d);
        acc.x += v.x * g; acc.y += v.y * g; acc.z += v.z * g; acc.w += v.w * g;
    }
    float inv = (hi > lo) ? (1.0f / (float)(hi - lo)) : 0.0f;
    float mx = fmaxf(acc.x * inv, 0.f), my = fmaxf(acc.y * inv, 0.f);
    float mz = fmaxf(acc.z * inv, 0.f), mw = fmaxf(acc.w * inv, 0.f);
    uint2 pk;
    pk.x = pack2bf(mx, my);
    pk.y = pack2bf(mz, mw);
    *(uint2*)(mreluOut + (long)t * D_C + d) = pk;
}

// ---------------------------------------------------------------------------
// K3: transform GEMM  out = feat + relu(m) @ twT + tb. One wave per 16x16 tile.
// ---------------------------------------------------------------------------
__global__ __launch_bounds__(256) void transform_kernel(
    const unsigned short* __restrict__ mrelu,     // [T, 512] bf16
    const unsigned int* __restrict__ bpack,       // [16][32][32][8] dwords
    const float* __restrict__ bias,               // [512]
    const float* __restrict__ feat,               // [T, 512] fp32
    float* __restrict__ out,                      // [T, 512] fp32
    int T) {
    int lane = threadIdx.x & 31;
    int wave = threadIdx.x >> 5;
    int w    = blockIdx.x * 8 + wave;
    int mt   = w >> 5;           // 32 N-tiles
    int nt   = w & 31;
    if (mt * 16 >= T) return;

    int r  = lane & 15;
    int hi = lane >> 4;
    const unsigned short* arow = mrelu + (long)(mt * 16 + r) * D_C;

    v8f c = {0.f, 0.f, 0.f, 0.f, 0.f, 0.f, 0.f, 0.f};
    for (int kk = 0; kk < 16; ++kk) {
        int K0 = kk * 32;
        v8u au;
#pragma unroll
        for (int p = 0; p < 4; ++p)
            au[p] = *(const unsigned int*)(arow + K0 + hi * 8 + 2 * p);
#pragma unroll
        for (int p = 0; p < 4; ++p)
            au[4 + p] = *(const unsigned int*)(arow + K0 + 16 + hi * 8 + 2 * p);
        v16bf a = __builtin_bit_cast(v16bf, au);
        v16bf b = __builtin_bit_cast(v16bf,
            *((const v8u*)bpack + (long)(kk * 32 + nt) * 32 + lane));
        c = __builtin_amdgcn_wmma_f32_16x16x32_bf16(
            false, a, false, b, (short)0, c, false, false);
    }

    int N = nt * 16 + r;
    float bn = bias[N];
#pragma unroll
    for (int q = 0; q < 8; ++q) {
        long M = (long)(mt * 16 + hi * 8 + q);
        out[M * D_C + N] = feat[M * D_C + N] + c[q] + bn;
    }
}

// ---------------------------------------------------------------------------
extern "C" void kernel_launch(void* const* d_in, const int* in_sizes, int n_in,
                              void* d_out, int out_size, void* d_ws, size_t ws_size,
                              hipStream_t stream) {
    const float* feat_obj = (const float*)d_in[0];
    const float* feat_reg = (const float*)d_in[1];
    const int*   r2o_tgt  = (const int*)d_in[2];
    const int*   r2o_src  = (const int*)d_in[3];
    const int*   o2r_tgt  = (const int*)d_in[4];
    const int*   o2r_src  = (const int*)d_in[5];
    const float* gw_r2o   = (const float*)d_in[6];
    const float* gb_r2o   = (const float*)d_in[7];
    const float* gw_o2r   = (const float*)d_in[8];
    const float* gb_o2r   = (const float*)d_in[9];
    const float* tw_r2o   = (const float*)d_in[10];
    const float* tb_r2o   = (const float*)d_in[11];
    const float* tw_o2r   = (const float*)d_in[12];
    const float* tb_o2r   = (const float*)d_in[13];

    float* out_obj = (float*)d_out;
    float* out_reg = out_obj + (size_t)N_OBJ_C * D_C;

    // ---- workspace layout (all 256B aligned; ~27 MB total) ----
    char* ws = (char*)d_ws;
    size_t off = 0;
    auto take = [&](size_t bytes) -> char* {
        char* p = ws + off;
        off += (bytes + 255) & ~(size_t)255;
        return p;
    };
    unsigned short* reluObj   = (unsigned short*)take((size_t)N_OBJ_C * D_C * 2);
    unsigned short* reluReg   = (unsigned short*)take((size_t)N_REG_C * D_C * 2);
    unsigned int*   gwpackR2O = (unsigned int*)take((size_t)32 * 8 * 32 * 8 * 4);
    unsigned int*   gwpackO2R = (unsigned int*)take((size_t)32 * 8 * 32 * 8 * 4);
    unsigned int*   twpackR2O = (unsigned int*)take((size_t)16 * 32 * 32 * 8 * 4);
    unsigned int*   twpackO2R = (unsigned int*)take((size_t)16 * 32 * 32 * 8 * 4);
    float*          gateR2O   = (float*)take((size_t)E_C * 4);
    float*          gateO2R   = (float*)take((size_t)E_C * 4);
    unsigned short* mreluObj  = (unsigned short*)take((size_t)N_OBJ_C * D_C * 2);
    unsigned short* mreluReg  = (unsigned short*)take((size_t)N_REG_C * D_C * 2);
    (void)ws_size;

    // ---- P1: relu + bf16 convert features ----
    {
        int n1 = N_OBJ_C * D_C, n2 = N_REG_C * D_C;
        relu_bf16_kernel<<<(n1 + 255) / 256, 256, 0, stream>>>(feat_obj, reluObj, n1);
        relu_bf16_kernel<<<(n2 + 255) / 256, 256, 0, stream>>>(feat_reg, reluReg, n2);
    }

    // ---- P2: pack gate & transform weights into B-fragment layout ----
    {
        int tot_gw = 32 * 8 * 32 * 8;     // ksteps * ntiles * lanes * dwords
        pack_w_kernel<<<(tot_gw + 255) / 256, 256, 0, stream>>>(gw_r2o, gwpackR2O, 1024, 8, tot_gw);
        pack_w_kernel<<<(tot_gw + 255) / 256, 256, 0, stream>>>(gw_o2r, gwpackO2R, 1024, 8, tot_gw);
        int tot_tw = 16 * 32 * 32 * 8;
        pack_w_kernel<<<(tot_tw + 255) / 256, 256, 0, stream>>>(tw_r2o, twpackR2O, 512, 32, tot_tw);
        pack_w_kernel<<<(tot_tw + 255) / 256, 256, 0, stream>>>(tw_o2r, twpackO2R, 512, 32, tot_tw);
    }

    // ---- K1: gates (WMMA bf16) ----
    {
        int tiles = E_C / 16;             // 8192 waves, 8 waves per block
        gate_kernel<<<tiles / 8, 256, 0, stream>>>(
            reluReg, reluObj, r2o_src, r2o_tgt, gwpackR2O, gb_r2o, gateR2O, E_C);
        gate_kernel<<<tiles / 8, 256, 0, stream>>>(
            reluObj, reluReg, o2r_src, o2r_tgt, gwpackO2R, gb_o2r, gateO2R, E_C);
    }

    // ---- K2: segmented mean + relu + bf16 ----
    agg_kernel<<<N_OBJ_C, 128, 0, stream>>>(feat_reg, r2o_src, r2o_tgt, gateR2O, mreluObj, E_C);
    agg_kernel<<<N_REG_C, 128, 0, stream>>>(feat_obj, o2r_src, o2r_tgt, gateO2R, mreluReg, E_C);

    // ---- K3: transform GEMM + bias + residual (WMMA bf16) ----
    {
        int waves_obj = (N_OBJ_C / 16) * 32;
        int waves_reg = (N_REG_C / 16) * 32;
        transform_kernel<<<waves_obj / 8, 256, 0, stream>>>(
            mreluObj, twpackR2O, tb_r2o, feat_obj, out_obj, N_OBJ_C);
        transform_kernel<<<waves_reg / 8, 256, 0, stream>>>(
            mreluReg, twpackO2R, tb_o2r, feat_reg, out_reg, N_REG_C);
    }
}
